// NLPPredictLayer_67044439490879
// MI455X (gfx1250) — matrix operand order repaired and checked by reading, not compiled, not run on hardware
//
#include <hip/hip_runtime.h>

// ---------------------------------------------------------------------------
// Problem constants (from reference): N=100000, E=500000, D_NLP=768, D_EMB=128
// ---------------------------------------------------------------------------
#define N_NODES 100000
#define N_EDGES 500000
#define D_NLP   768
#define D_EMB   128

typedef __attribute__((ext_vector_type(8)))  float   v8f;
typedef __attribute__((ext_vector_type(16))) __bf16  v16bf;
typedef __attribute__((ext_vector_type(8)))  __bf16  v8bf;

// ---------------------------------------------------------------------------
// Kernel 1: init the global min cell
// ---------------------------------------------------------------------------
__global__ void init_min_kernel(int* __restrict__ minv) {
    *minv = 0x7fffffff;
}

// ---------------------------------------------------------------------------
// Kernel 2: m = min over edges of max(e0, e1)
// ---------------------------------------------------------------------------
__global__ __launch_bounds__(256) void edge_min_kernel(const int* __restrict__ ei,
                                                       int* __restrict__ minv) {
    __shared__ int red[256];
    int m = 0x7fffffff;
    for (int e = blockIdx.x * 256 + threadIdx.x; e < N_EDGES; e += gridDim.x * 256) {
        int a = ei[e];
        int b = ei[N_EDGES + e];
        int mx = a > b ? a : b;
        m = mx < m ? mx : m;
    }
    red[threadIdx.x] = m;
    __syncthreads();
    #pragma unroll
    for (int s = 128; s > 0; s >>= 1) {
        if (threadIdx.x < (unsigned)s) {
            int o = red[threadIdx.x + s];
            if (o < red[threadIdx.x]) red[threadIdx.x] = o;
        }
        __syncthreads();
    }
    if (threadIdx.x == 0) atomicMin(minv, red[0]);
}

// ---------------------------------------------------------------------------
// Kernel 3: one-time w0 (128x768 f32) -> bf16 conversion (L2-resident reuse)
// ---------------------------------------------------------------------------
__global__ __launch_bounds__(256) void convert_w0_kernel(const float* __restrict__ w0,
                                                         __bf16* __restrict__ w0b) {
    int i = blockIdx.x * 256 + threadIdx.x;
    if (i < D_EMB * D_NLP) w0b[i] = (__bf16)w0[i];
}

// ---------------------------------------------------------------------------
// Kernel 4: per-node  s_a[n] = z[n].w1a,  s_b[n] = z[n].w1b
// One wave32 per node: float4 per lane (fully coalesced b128), shfl reduce.
// ---------------------------------------------------------------------------
__global__ __launch_bounds__(256) void node_dot_kernel(const float* __restrict__ z,
                                                       const float* __restrict__ w1,
                                                       float* __restrict__ s_a,
                                                       float* __restrict__ s_b) {
    const int wave = threadIdx.x >> 5;
    const int lane = threadIdx.x & 31;
    const int node = blockIdx.x * 8 + wave;
    if (node >= N_NODES) return;

    float4 zv = ((const float4*)(z + (size_t)node * D_EMB))[lane];
    float4 wa = ((const float4*)(w1))[lane];
    float4 wb = ((const float4*)(w1 + D_EMB))[lane];

    float da = zv.x * wa.x + zv.y * wa.y + zv.z * wa.z + zv.w * wa.w;
    float db = zv.x * wb.x + zv.y * wb.y + zv.z * wb.z + zv.w * wb.w;

    #pragma unroll
    for (int off = 16; off > 0; off >>= 1) {
        da += __shfl_xor(da, off, 32);
        db += __shfl_xor(db, off, 32);
    }
    if (lane == 0) { s_a[node] = da; s_b[node] = db; }
}

// ---------------------------------------------------------------------------
// Kernel 5: t[n] = dot(PReLU(z0[n] @ w0^T), w1c)  via bf16 WMMA, f32 accum.
//
// Block = 256 threads = 8 wave32s, covers ROWS=64 node rows.
//   - z0 rows staged to LDS as bf16 (padded stride -> conflict-free b128 A frags)
//   - wave w owns 16 output columns (8*16 = 128 = D_EMB), 4 row sub-tiles
//   - K = 768 in 24 steps of V_WMMA_F32_16X16X32_BF16; B frag reused 4x
// Fragment layouts per CDNA5 ISA 7.12.2:
//   A(16x32 bf16): lane row = lane&15; lo half K {kb+0..7, kb+16..23},
//                  hi half K {kb+8..15, kb+24..31}  -> two 16B chunks
//   B(32x16 bf16): lane col = lane&15; lo half K {kb+0..7, kb+16..23}, hi half {kb+8..15, kb+24..31}
//                  -> two 16B chunks of w0 row (w0^T column)
//   C/D(16x16 f32): VGPR v, lanes0-15 M=v, lanes16-31 M=v+8, N=lane&15
// ---------------------------------------------------------------------------
#define ROWS        64
#define Z0S_STRIDE  (D_NLP + 8)   // 776 bf16 -> 388 dwords, 388 % 64 = 4 (no conflicts)
#define HP_STRIDE   (D_EMB + 1)   // 129

__global__ __launch_bounds__(256) void gemm_prelu_dot_kernel(
    const float*  __restrict__ z0,
    const __bf16* __restrict__ w0b,
    const float*  __restrict__ w1c,
    const float*  __restrict__ prelu_a,
    float*        __restrict__ t) {

    __shared__ union {
        __bf16 z0s[ROWS][Z0S_STRIDE];   // 99328 B, live during MAC loop
        float  hp[ROWS][HP_STRIDE];     // 33024 B, live after MAC loop
    } smem;

    const int rowBase = blockIdx.x * ROWS;

    // ---- stage z0 tile -> bf16 LDS (coalesced float4 global loads) ----
    for (int i = threadIdx.x; i < ROWS * (D_NLP / 4); i += 256) {
        int r  = i / (D_NLP / 4);
        int c4 = i - r * (D_NLP / 4);
        int node = rowBase + r;
        float4 v = make_float4(0.f, 0.f, 0.f, 0.f);
        if (node < N_NODES)
            v = ((const float4*)(z0 + (size_t)node * D_NLP))[c4];
        __bf16* dst = &smem.z0s[r][c4 * 4];
        dst[0] = (__bf16)v.x; dst[1] = (__bf16)v.y;
        dst[2] = (__bf16)v.z; dst[3] = (__bf16)v.w;
    }
    __syncthreads();

    const int wave = threadIdx.x >> 5;
    const int lane = threadIdx.x & 31;
    const int l15  = lane & 15;
    const int hi   = lane >> 4;            // 0: lanes 0-15, 1: lanes 16-31
    const int colBase = wave * 16;

    v8f acc[4] = {};                        // 4 row sub-tiles x (16x16 f32)

    const __bf16* wrow = w0b + (size_t)(colBase + l15) * D_NLP;

    for (int kb = 0; kb < D_NLP; kb += 32) {
        // B fragment: K {kb+0..7, kb+16..23} (+8 hi) of w0 row (L2-hot)
        const v8bf* bp = (const v8bf*)(wrow + kb + hi * 8);
        v8bf b0 = bp[0];
        v8bf b1v = bp[2];
        v16bf bfrag = __builtin_shufflevector(b0, b1v,
            0, 1, 2, 3, 4, 5, 6, 7, 8, 9, 10, 11, 12, 13, 14, 15);

        #pragma unroll
        for (int rt = 0; rt < 4; ++rt) {
            const __bf16* arow = &smem.z0s[rt * 16 + l15][0];
            v8bf a0 = *(const v8bf*)(arow + kb + hi * 8);        // K kb..kb+7 (+8 hi)
            v8bf a1 = *(const v8bf*)(arow + kb + 16 + hi * 8);   // K kb+16..23 (+8 hi)
            v16bf afrag = __builtin_shufflevector(a0, a1,
                0, 1, 2, 3, 4, 5, 6, 7, 8, 9, 10, 11, 12, 13, 14, 15);

            acc[rt] = __builtin_amdgcn_wmma_f32_16x16x32_bf16(
                /*neg_a=*/false, afrag, /*neg_b=*/false, bfrag,
                /*c_mod=*/(short)0, acc[rt],
                /*reuse_a=*/false, /*reuse_b=*/false);
        }
    }

    __syncthreads();   // all waves done reading z0s; LDS is reused as hp

    const float ap = prelu_a[0];
    const int   col = colBase + l15;
    const float wc  = w1c[col];

    #pragma unroll
    for (int rt = 0; rt < 4; ++rt) {
        #pragma unroll
        for (int v = 0; v < 8; ++v) {
            int row = rt * 16 + v + hi * 8;
            float val = acc[rt][v];
            float x = (val >= 0.f) ? val : ap * val;   // PReLU
            smem.hp[row][col] = x * wc;
        }
    }
    __syncthreads();

    // reduce 128 columns per row -> t[node]
    if (threadIdx.x < ROWS) {
        int node = rowBase + threadIdx.x;
        float s = 0.f;
        #pragma unroll 8
        for (int c = 0; c < D_EMB; ++c) s += smem.hp[threadIdx.x][c];
        if (node < N_NODES) t[node] = s;
    }
}

// ---------------------------------------------------------------------------
// Kernel 6: per-edge gather of the three L2-hot 400 KB tables
// ---------------------------------------------------------------------------
__global__ __launch_bounds__(256) void edge_out_kernel(
    const int*   __restrict__ ei,
    const float* __restrict__ s_a,
    const float* __restrict__ s_b,
    const float* __restrict__ t,
    const float* __restrict__ b1,
    const int*   __restrict__ minv,
    float*       __restrict__ out) {
    int e = blockIdx.x * 256 + threadIdx.x;
    if (e >= N_EDGES) return;
    int a = ei[e];
    int b = ei[N_EDGES + e];
    int q = (a > b ? a : b) - *minv;
    out[e] = s_a[a] + s_b[b] + t[q] + b1[0];
}

// ---------------------------------------------------------------------------
// Launch: inputs in setup_inputs() order:
//   d_in[0]=z (N,128) f32   d_in[1]=z0 (N,768) f32   d_in[2]=edge_index (2,E) i32
//   d_in[3]=w0 (128,768) f32  d_in[4]=prelu_a (1,) f32
//   d_in[5]=w1 (1,384) f32    d_in[6]=b1 (1,) f32
// Workspace layout (bytes): [0:4] min | [256: +196608] w0 bf16 |
//   then s_a (400000) | s_b (400000) | t (400000)   (~1.4 MB total)
// ---------------------------------------------------------------------------
extern "C" void kernel_launch(void* const* d_in, const int* in_sizes, int n_in,
                              void* d_out, int out_size, void* d_ws, size_t ws_size,
                              hipStream_t stream) {
    const float* z  = (const float*)d_in[0];
    const float* z0 = (const float*)d_in[1];
    const int*   ei = (const int*)d_in[2];
    const float* w0 = (const float*)d_in[3];
    const float* pa = (const float*)d_in[4];
    const float* w1 = (const float*)d_in[5];
    const float* b1 = (const float*)d_in[6];
    float* out = (float*)d_out;

    char* ws = (char*)d_ws;
    int*    minv = (int*)ws;                                   // 4 B (padded to 256)
    __bf16* w0b  = (__bf16*)(ws + 256);                        // 196608 B
    float*  s_a  = (float*)(ws + 256 + (size_t)D_EMB * D_NLP * 2);
    float*  s_b  = s_a + N_NODES;
    float*  t    = s_b + N_NODES;

    init_min_kernel<<<1, 1, 0, stream>>>(minv);
    edge_min_kernel<<<512, 256, 0, stream>>>(ei, minv);
    convert_w0_kernel<<<(D_EMB * D_NLP + 255) / 256, 256, 0, stream>>>(w0, w0b);
    node_dot_kernel<<<(N_NODES + 7) / 8, 256, 0, stream>>>(z, w1, s_a, s_b);
    gemm_prelu_dot_kernel<<<(N_NODES + ROWS - 1) / ROWS, 256, 0, stream>>>(
        z0, w0b, w1 + 2 * D_EMB, pa, t);
    edge_out_kernel<<<(N_EDGES + 255) / 256, 256, 0, stream>>>(
        ei, s_a, s_b, t, b1, minv, out);
}
